// Cross_48060684042859
// MI455X (gfx1250) — matrix operand order repaired and checked by reading, not compiled, not run on hardware
//
#include <hip/hip_runtime.h>
#include <hip/hip_bf16.h>

// Problem dims (fixed by the reference)
#define H_ 8
#define N_ 4096
#define M_ 4096
#define F_ 1024
#define D_ 256

typedef __attribute__((ext_vector_type(16))) __bf16 v16bf;
typedef __attribute__((ext_vector_type(8)))  __bf16 v8bf;
typedef __attribute__((ext_vector_type(4)))  __bf16 v4bf;
typedef __attribute__((ext_vector_type(8)))  float  v8f;
typedef __attribute__((ext_vector_type(4)))  unsigned int u32x4;
typedef __attribute__((ext_vector_type(8)))  unsigned int u32x8;

// LDS row strides (elements) after TDM padding (16B pad per row => no bank
// conflicts for 16-lane ds_load_b128 fragment reads)
#define KBUF_LD 264   // 256 data + 8 pad elems -> 528B row stride
#define VBUF_LD 40    // 32 data + 8 pad elems  -> 80B row stride

// ---------------------------------------------------------------------------
// WMMA helper (CDNA5 v_wmma_f32_16x16x32_bf16, D = A(16x32) * B(32x16) + C)
// ---------------------------------------------------------------------------
__device__ __forceinline__ v8f wmma_bf16(v16bf a, v16bf b, v8f c) {
  return __builtin_amdgcn_wmma_f32_16x16x32_bf16(
      false, a, false, b, (short)0, c, false, false);
}

// A-fragment (16x32 bf16, row-major source): two contiguous 16B loads/lane.
__device__ __forceinline__ v16bf load_a_frag(const __bf16* tile, int ld, int lane) {
  const __bf16* p = tile + (size_t)(lane & 15) * ld + ((lane & 16) ? 16 : 0);
  union { v16bf v; v8bf h[2]; } u;
  u.h[0] = *(const v8bf*)(p);
  u.h[1] = *(const v8bf*)(p + 8);
  return u.v;
}

// B-fragment (32x16 bf16) from row-major B^T: two contiguous 16B loads/lane.
__device__ __forceinline__ v16bf load_bt_frag(const __bf16* tile, int ld, int lane) {
  const __bf16* p = tile + (size_t)(lane & 15) * ld + ((lane & 16) ? 16 : 0);
  union { v16bf v; v8bf h[2]; } u;
  u.h[0] = *(const v8bf*)(p);
  u.h[1] = *(const v8bf*)(p + 8);
  return u.v;
}

// ---------------------------------------------------------------------------
// TDM: issue tensor_load_to_lds for a 2D tile (bf16 elements).
// D# group0: count=1, lds_addr, 57b global byte address, type=2 ("image").
// D# group1: data_size=1 (2B), tensor dims / tile dims / dim0 stride, and
// optional LDS padding (pad_amount dwords after every 2^(pi+1) dwords).
// ---------------------------------------------------------------------------
__device__ __forceinline__ void tdm_load_2d_bf16(
    unsigned lds_off, const void* gaddr,
    unsigned tensor_d0, unsigned tensor_d1,
    unsigned tile_d0, unsigned tile_d1,
    unsigned stride_d0_elems,
    unsigned pad_interval_code, unsigned pad_amount_code) {
  unsigned long long ga = (unsigned long long)gaddr;
  u32x4 g0;
  g0[0] = 1u;                                            // count=1 (valid)
  g0[1] = lds_off;                                       // lds_addr (bytes)
  g0[2] = (unsigned)(ga & 0xffffffffu);                  // global_addr lo
  g0[3] = (unsigned)((ga >> 32) & 0x01ffffffu) | (2u << 30);  // addr hi | type=2
  u32x8 g1;
  g1[0] = (1u << 16)                                     // data_size = 2 bytes
        | (1u << 20)                                     // pad_enable
        | (pad_interval_code << 22) | (pad_amount_code << 25);
  g1[1] = (tensor_d0 & 0xffffu) << 16;                   // tensor_dim0[15:0]
  g1[2] = ((tensor_d0 >> 16) & 0xffffu) | ((tensor_d1 & 0xffffu) << 16);
  g1[3] = ((tensor_d1 >> 16) & 0xffffu) | ((tile_d0 & 0xffffu) << 16);
  g1[4] = (tile_d1 & 0xffffu);                           // tile_dim2 = 0 (2D)
  g1[5] = stride_d0_elems;                               // dim0 stride lo32
  g1[6] = 0;                                             // stride hi / dim1 stride
  g1[7] = 0;
  asm volatile("tensor_load_to_lds %0, %1" :: "s"(g0), "s"(g1) : "memory");
}

// ---------------------------------------------------------------------------
// Kernel: f32 -> bf16 convert (vectorized x4)
// ---------------------------------------------------------------------------
__global__ void cvt_f32_bf16(const float* __restrict__ in, __bf16* __restrict__ out,
                             size_t n) {
  size_t i = ((size_t)blockIdx.x * blockDim.x + threadIdx.x) * 4;
  if (i + 3 < n) {
    float4 f = *(const float4*)(in + i);
    v4bf o;
    o[0] = (__bf16)f.x; o[1] = (__bf16)f.y; o[2] = (__bf16)f.z; o[3] = (__bf16)f.w;
    *(v4bf*)(out + i) = o;
  }
}

// ---------------------------------------------------------------------------
// Kernel: batched transpose + convert: in f32 [R][C] -> out bf16 [C][R]
// ---------------------------------------------------------------------------
__global__ void transpose_cvt(const float* __restrict__ in, __bf16* __restrict__ out,
                              int R, int C) {
  __shared__ float tile[32][33];
  const float* inb  = in  + (size_t)blockIdx.z * R * C;
  __bf16*      outb = out + (size_t)blockIdx.z * R * C;
  int c0 = blockIdx.x * 32, r0 = blockIdx.y * 32;
  int tx = threadIdx.x, ty = threadIdx.y;
  for (int i = ty; i < 32; i += 8)
    tile[i][tx] = inb[(size_t)(r0 + i) * C + (c0 + tx)];
  __syncthreads();
  for (int i = ty; i < 32; i += 8)
    outb[(size_t)(c0 + i) * R + (r0 + tx)] = (__bf16)tile[tx][i];
}

// ---------------------------------------------------------------------------
// Kernel: QKV projections. blockIdx.z: 0=q (from this), 1=k, 2=v (from that).
// 8 waves; workgroup tile 64x256, wave 32x64. v stored transposed vT[h][d][m].
// Branch-free epilogue via uniform (base, row-stride, col-stride).
// ---------------------------------------------------------------------------
__global__ __launch_bounds__(256)
void proj_gemm(const __bf16* __restrict__ thisb, const __bf16* __restrict__ thatb,
               const __bf16* __restrict__ WqT, const __bf16* __restrict__ WkT,
               const __bf16* __restrict__ WvT,
               const float* __restrict__ bq, const float* __restrict__ bk,
               const float* __restrict__ bv,
               __bf16* __restrict__ q, __bf16* __restrict__ k,
               __bf16* __restrict__ vT) {
  const int lane = threadIdx.x & 31;
  const int w    = threadIdx.x >> 5;
  const int wrow = w >> 2, wcol = w & 3;
  const int h = blockIdx.y;
  const int z = blockIdx.z;
  const int row0 = blockIdx.x * 64 + wrow * 32;
  const int col0 = wcol * 64;

  const __bf16* A  = (z == 0) ? thisb : thatb;
  const __bf16* WT = ((z == 0) ? WqT : (z == 1) ? WkT : WvT) + (size_t)h * D_ * F_;
  const float* bias = ((z == 0) ? bq : (z == 1) ? bk : bv) + h * D_;

  // Uniform destination mapping: addr = base + row*rstr + col*cstr
  __bf16* dst; size_t rstr, cstr;
  if (z == 0)      { dst = q  + (size_t)h * N_ * D_; rstr = D_; cstr = 1; }
  else if (z == 1) { dst = k  + (size_t)h * M_ * D_; rstr = D_; cstr = 1; }
  else             { dst = vT + (size_t)h * D_ * M_; rstr = 1;  cstr = M_; }

  v8f acc[2][4];
#pragma unroll
  for (int r = 0; r < 2; ++r)
#pragma unroll
    for (int c = 0; c < 4; ++c) acc[r][c] = v8f{};

  for (int kk = 0; kk < F_; kk += 32) {
    v16bf af[2], bf[4];
#pragma unroll
    for (int r = 0; r < 2; ++r)
      af[r] = load_a_frag(A + (size_t)(row0 + 16 * r) * F_ + kk, F_, lane);
#pragma unroll
    for (int c = 0; c < 4; ++c)
      bf[c] = load_bt_frag(WT + (size_t)(col0 + 16 * c) * F_ + kk, F_, lane);
#pragma unroll
    for (int r = 0; r < 2; ++r)
#pragma unroll
      for (int c = 0; c < 4; ++c)
        acc[r][c] = wmma_bf16(af[r], bf[c], acc[r][c]);
  }

  const int rsel = (lane & 16) ? 8 : 0;
#pragma unroll
  for (int r = 0; r < 2; ++r)
#pragma unroll
    for (int c = 0; c < 4; ++c) {
      const int col = col0 + 16 * c + (lane & 15);
      const float b = bias[col];
      __bf16* dcol = dst + (size_t)col * cstr + (size_t)(row0 + 16 * r + rsel) * rstr;
#pragma unroll
      for (int i = 0; i < 8; ++i)
        dcol[(size_t)i * rstr] = (__bf16)(acc[r][c][i] + b);
    }
}

// ---------------------------------------------------------------------------
// Kernel: flash attention with TDM-staged k/v tiles in LDS.
// 8 waves/block, each wave owns 16 query rows x D=256; block shares head h.
// Per 32-column step: wave 0 issues two tensor_load_to_lds (k tile 32x256,
// v tile 256x32 from vT) with padded LDS rows, s_wait_tensorcnt, barrier;
// then 16 score WMMAs (LDS B-frags), mask, online softmax (shfl_xor +
// v_exp_f32), p-tile LDS bounce, 16 p@v WMMAs (LDS B-frags).
// ---------------------------------------------------------------------------
__global__ __launch_bounds__(256)
void attn_kernel(const __bf16* __restrict__ q, const __bf16* __restrict__ k,
                 const __bf16* __restrict__ vT, const unsigned char* __restrict__ mask,
                 __bf16* __restrict__ cat) {
  __shared__ __align__(16) __bf16 kbuf[32 * KBUF_LD];    // 16,896 B
  __shared__ __align__(16) __bf16 vbuf[256 * VBUF_LD];   // 20,480 B
  __shared__ __align__(16) __bf16 plds[8][16][32];       //  8,192 B
  const int lane = threadIdx.x & 31;
  const int w    = threadIdx.x >> 5;
  const int h    = blockIdx.y;
  const int q0   = blockIdx.x * 128 + w * 16;
  const __bf16* qh = q  + (size_t)h * N_ * D_;
  const __bf16* kh = k  + (size_t)h * M_ * D_;
  const __bf16* vh = vT + (size_t)h * D_ * M_;
  const bool use_mask = (h >= 4);
  const int rsel = (lane & 16) ? 8 : 0;
  const int rowbase = q0 + rsel;
  const unsigned kbuf_off = (unsigned)(unsigned long long)(const void*)&kbuf[0];
  const unsigned vbuf_off = (unsigned)(unsigned long long)(const void*)&vbuf[0];

  // q fragments, register resident for the whole kernel (64 VGPRs)
  v16bf qf[8];
#pragma unroll
  for (int t = 0; t < 8; ++t)
    qf[t] = load_a_frag(qh + (size_t)q0 * D_ + t * 32, D_, lane);

  v8f rv[16];
#pragma unroll
  for (int d = 0; d < 16; ++d) rv[d] = v8f{};
  float mrun[8], lrun[8];
#pragma unroll
  for (int i = 0; i < 8; ++i) { mrun[i] = -3.0e38f; lrun[i] = 0.0f; }

  for (int m0 = 0; m0 < M_; m0 += 32) {
    // ---- TDM: stage k[m0..m0+31][0..255] and vT[0..255][m0..m0+31] into LDS
    if (w == 0) {
      // k tile: 32 rows x 256 elems; row=512B -> pad code 6 (128 dw), +16B (code 3)
      tdm_load_2d_bf16(kbuf_off, kh + (size_t)m0 * D_,
                       D_, M_, D_, 32, D_, 6u, 3u);
      // v tile: 256 rows x 32 elems; row=64B -> pad code 3 (16 dw), +16B (code 3)
      tdm_load_2d_bf16(vbuf_off, vh + m0,
                       M_, D_, 32, D_, M_, 3u, 3u);
      __builtin_amdgcn_s_wait_tensorcnt(0);
    }
    __syncthreads();   // tiles visible to all waves

    // ---- scores: two 16x16 tiles covering columns m0..m0+31 (B-frags from LDS)
    v8f s0 = v8f{}, s1 = v8f{};
#pragma unroll
    for (int t = 0; t < 8; ++t) {
      const int koff = t * 32 + ((lane & 16) ? 16 : 0);
      const __bf16* p0 = &kbuf[(lane & 15) * KBUF_LD + koff];
      const __bf16* p1 = &kbuf[((lane & 15) + 16) * KBUF_LD + koff];
      union { v16bf v; v8bf h2[2]; } u0, u1;
      u0.h2[0] = *(const v8bf*)p0; u0.h2[1] = *(const v8bf*)(p0 + 8);
      u1.h2[0] = *(const v8bf*)p1; u1.h2[1] = *(const v8bf*)(p1 + 8);
      s0 = wmma_bf16(qf[t], u0.v, s0);
      s1 = wmma_bf16(qf[t], u1.v, s1);
    }
    // ---- additive mask for heads >= 4
    if (use_mask) {
      const int col = m0 + (lane & 15);
#pragma unroll
      for (int i = 0; i < 8; ++i) {
        const size_t mrow = (size_t)(rowbase + i) * M_;
        if (!mask[mrow + col])      s0[i] -= 1e9f;
        if (!mask[mrow + col + 16]) s1[i] -= 1e9f;
      }
    }
    // ---- online softmax (row stats live per 16-lane half; xor 1/2/4/8)
    float alpha[8];
#pragma unroll
    for (int i = 0; i < 8; ++i) {
      float tm = fmaxf(s0[i], s1[i]);
      tm = fmaxf(tm, __shfl_xor(tm, 1, 32));
      tm = fmaxf(tm, __shfl_xor(tm, 2, 32));
      tm = fmaxf(tm, __shfl_xor(tm, 4, 32));
      tm = fmaxf(tm, __shfl_xor(tm, 8, 32));
      const float mnew = fmaxf(mrun[i], tm);
      alpha[i] = __expf(mrun[i] - mnew);
      mrun[i] = mnew;
      const float p0 = __expf(s0[i] - mnew);
      const float p1 = __expf(s1[i] - mnew);
      s0[i] = p0; s1[i] = p1;
      float rs = p0 + p1;
      rs += __shfl_xor(rs, 1, 32);
      rs += __shfl_xor(rs, 2, 32);
      rs += __shfl_xor(rs, 4, 32);
      rs += __shfl_xor(rs, 8, 32);
      lrun[i] = lrun[i] * alpha[i] + rs;
    }
#pragma unroll
    for (int d = 0; d < 16; ++d)
#pragma unroll
      for (int i = 0; i < 8; ++i) rv[d][i] *= alpha[i];

    // ---- C-layout -> A-fragment re-layout via wave-private LDS tile
    {
      const int cl = lane & 15;
#pragma unroll
      for (int i = 0; i < 8; ++i) {
        plds[w][rsel + i][cl]      = (__bf16)s0[i];
        plds[w][rsel + i][cl + 16] = (__bf16)s1[i];
      }
    }
    asm volatile("s_wait_dscnt 0" ::: "memory");  // same-wave LDS RAW fence
    v16bf pf;
    {
      const __bf16* p = &plds[w][lane & 15][(lane & 16) ? 16 : 0];
      union { v16bf v; v8bf h2[2]; } u;
      u.h2[0] = *(const v8bf*)(p);
      u.h2[1] = *(const v8bf*)(p + 8);
      pf = u.v;
    }
    // ---- r += p @ v  (B^T rows = vbuf rows, padded stride)
#pragma unroll
    for (int dt = 0; dt < 16; ++dt) {
      const __bf16* p = &vbuf[(dt * 16 + (lane & 15)) * VBUF_LD + ((lane & 16) ? 16 : 0)];
      union { v16bf v; v8bf h2[2]; } u;
      u.h2[0] = *(const v8bf*)p;
      u.h2[1] = *(const v8bf*)(p + 8);
      rv[dt] = wmma_bf16(pf, u.v, rv[dt]);
    }
    __syncthreads();   // all reads done before next TDM overwrites the tiles
  }

  // ---- finalize: divide by l, write concatenated heads (bf16)
#pragma unroll
  for (int i = 0; i < 8; ++i) lrun[i] = 1.0f / lrun[i];
#pragma unroll
  for (int d = 0; d < 16; ++d) {
    const int col = h * D_ + d * 16 + (lane & 15);
#pragma unroll
    for (int i = 0; i < 8; ++i) {
      const int row = rowbase + i;
      cat[(size_t)row * (2 * F_) + col] = (__bf16)(rv[d][i] * lrun[i]);
    }
  }
}

// ---------------------------------------------------------------------------
// Kernel: output GEMM  out = cat[4096x2048] @ Wo[2048x1024] + bo + this (f32)
// ---------------------------------------------------------------------------
__global__ __launch_bounds__(256)
void out_gemm(const __bf16* __restrict__ cat, const __bf16* __restrict__ WoT,
              const float* __restrict__ bo, const float* __restrict__ resid,
              float* __restrict__ out) {
  const int lane = threadIdx.x & 31;
  const int w    = threadIdx.x >> 5;
  const int wrow = w >> 2, wcol = w & 3;
  const int row0 = blockIdx.x * 64 + wrow * 32;
  const int col0 = blockIdx.y * 256 + wcol * 64;
  const int K = 2 * F_;

  v8f acc[2][4];
#pragma unroll
  for (int r = 0; r < 2; ++r)
#pragma unroll
    for (int c = 0; c < 4; ++c) acc[r][c] = v8f{};

  for (int kk = 0; kk < K; kk += 32) {
    v16bf af[2], bf[4];
#pragma unroll
    for (int r = 0; r < 2; ++r)
      af[r] = load_a_frag(cat + (size_t)(row0 + 16 * r) * K + kk, K, lane);
#pragma unroll
    for (int c = 0; c < 4; ++c)
      bf[c] = load_bt_frag(WoT + (size_t)(col0 + 16 * c) * K + kk, K, lane);
#pragma unroll
    for (int r = 0; r < 2; ++r)
#pragma unroll
      for (int c = 0; c < 4; ++c)
        acc[r][c] = wmma_bf16(af[r], bf[c], acc[r][c]);
  }

  const int rsel = (lane & 16) ? 8 : 0;
#pragma unroll
  for (int r = 0; r < 2; ++r)
#pragma unroll
    for (int c = 0; c < 4; ++c) {
      const int col = col0 + 16 * c + (lane & 15);
      const float b = bo[col];
#pragma unroll
      for (int i = 0; i < 8; ++i) {
        const int row = row0 + 16 * r + i + rsel;
        out[(size_t)row * F_ + col] = acc[r][c][i] + b + resid[(size_t)row * F_ + col];
      }
    }
}

// ---------------------------------------------------------------------------
// Host-side launcher
// ---------------------------------------------------------------------------
extern "C" void kernel_launch(void* const* d_in, const int* in_sizes, int n_in,
                              void* d_out, int out_size, void* d_ws, size_t ws_size,
                              hipStream_t stream) {
  const float* d_this = (const float*)d_in[0];
  const float* d_that = (const float*)d_in[1];
  const unsigned char* d_mask = (const unsigned char*)d_in[2];  // JAX bool = 1 byte
  const float* d_Wq = (const float*)d_in[3];
  const float* d_bq = (const float*)d_in[4];
  const float* d_Wk = (const float*)d_in[5];
  const float* d_bk = (const float*)d_in[6];
  const float* d_Wv = (const float*)d_in[7];
  const float* d_bv = (const float*)d_in[8];
  const float* d_Wo = (const float*)d_in[9];
  const float* d_bo = (const float*)d_in[10];
  float* out = (float*)d_out;

  // Workspace carve-out (~96 MB total)
  char* ws = (char*)d_ws;
  size_t off = 0;
  auto carve = [&](size_t bytes) { char* p = ws + off; off += bytes; return p; };
  __bf16* thisb = (__bf16*)carve((size_t)N_ * F_ * 2);
  __bf16* thatb = (__bf16*)carve((size_t)M_ * F_ * 2);
  __bf16* WqT   = (__bf16*)carve((size_t)H_ * D_ * F_ * 2);
  __bf16* WkT   = (__bf16*)carve((size_t)H_ * D_ * F_ * 2);
  __bf16* WvT   = (__bf16*)carve((size_t)H_ * D_ * F_ * 2);
  __bf16* WoT   = (__bf16*)carve((size_t)F_ * 2 * F_ * 2);
  __bf16* qb    = (__bf16*)carve((size_t)H_ * N_ * D_ * 2);
  __bf16* kb    = (__bf16*)carve((size_t)H_ * M_ * D_ * 2);
  __bf16* vTb   = (__bf16*)carve((size_t)H_ * D_ * M_ * 2);
  __bf16* catb  = (__bf16*)carve((size_t)N_ * 2 * F_ * 2);

  cvt_f32_bf16<<<(N_ * F_ / 4) / 256, 256, 0, stream>>>(d_this, thisb, (size_t)N_ * F_);
  cvt_f32_bf16<<<(M_ * F_ / 4) / 256, 256, 0, stream>>>(d_that, thatb, (size_t)M_ * F_);

  transpose_cvt<<<dim3(D_ / 32, F_ / 32, H_), dim3(32, 8), 0, stream>>>(d_Wq, WqT, F_, D_);
  transpose_cvt<<<dim3(D_ / 32, F_ / 32, H_), dim3(32, 8), 0, stream>>>(d_Wk, WkT, F_, D_);
  transpose_cvt<<<dim3(D_ / 32, F_ / 32, H_), dim3(32, 8), 0, stream>>>(d_Wv, WvT, F_, D_);
  transpose_cvt<<<dim3(F_ / 32, (2 * F_) / 32, 1), dim3(32, 8), 0, stream>>>(d_Wo, WoT, 2 * F_, F_);

  proj_gemm<<<dim3(N_ / 64, H_, 3), 256, 0, stream>>>(
      thisb, thatb, WqT, WkT, WvT, d_bq, d_bk, d_bv, qb, kb, vTb);

  attn_kernel<<<dim3(N_ / 128, H_), 256, 0, stream>>>(qb, kb, vTb, d_mask, catb);

  out_gemm<<<dim3(N_ / 64, F_ / 256), 256, 0, stream>>>(catb, WoT, d_bo, d_this, out);
}
